// Spell2Vec_78314433675757
// MI455X (gfx1250) — matrix operand order repaired and checked by reading, not compiled, not run on hardware
//
#include <hip/hip_runtime.h>
#include <hip/hip_bf16.h>
#include <math.h>

#define BATCH 2048
#define HDIM  512
#define EDIM  64
#define TSTEPS 20
#define G3H   1536      // 3*H
#define SPW   21        // spelling record width (T + 1)
#define MTILE 32        // batch rows per block (and per wave)
#define LDSH_STRIDE 520 // 512 + 8 bf16 pad
#define LDSX_STRIDE 72  // 64 + 8 bf16 pad

typedef __attribute__((ext_vector_type(16))) __bf16 v16bf;
typedef __attribute__((ext_vector_type(8)))  float  v8f;

union Frag {
    v16bf v;
    uint4 q[2];
};

#define WMMA_BF16(A, B, C) \
    __builtin_amdgcn_wmma_f32_16x16x32_bf16(false, (A).v, false, (B).v, (short)0, (C), false, false)

// ---------------------------------------------------------------------------
// prep: f32 -> bf16 weight/emb conversion + zero h ping-pong buffers
// ---------------------------------------------------------------------------
__global__ __launch_bounds__(256) void s2v_prep(
    const float* __restrict__ Whh_f, const float* __restrict__ Wih_f,
    const float* __restrict__ emb_f,
    __bf16* __restrict__ Whh, __bf16* __restrict__ Wih, __bf16* __restrict__ embb,
    float* __restrict__ h0f, __bf16* __restrict__ h0b,
    float* __restrict__ h1f, __bf16* __restrict__ h1b)
{
    const int idx = blockIdx.x * blockDim.x + threadIdx.x;
    const int stride = gridDim.x * blockDim.x;
    for (int i = idx; i < G3H * HDIM; i += stride) Whh[i] = (__bf16)Whh_f[i];
    for (int i = idx; i < G3H * EDIM; i += stride) Wih[i] = (__bf16)Wih_f[i];
    for (int i = idx; i < 128 * EDIM; i += stride) embb[i] = (__bf16)emb_f[i];
    for (int i = idx; i < BATCH * HDIM; i += stride) {
        h0f[i] = 0.0f; h1f[i] = 0.0f;
        h0b[i] = (__bf16)0.0f; h1b[i] = (__bf16)0.0f;
    }
}

// ---------------------------------------------------------------------------
// build_perm: stable counting sort of lengths descending == argsort(-len)
// ---------------------------------------------------------------------------
__global__ __launch_bounds__(256) void s2v_perm(
    const int* __restrict__ data, const int* __restrict__ spell,
    int* __restrict__ perm)
{
    __shared__ int slen[BATCH];
    __shared__ int hist[SPW];
    __shared__ int start[SPW];
    const int tid = threadIdx.x;
    if (tid < SPW) hist[tid] = 0;
    __syncthreads();
    for (int i = tid; i < BATCH; i += blockDim.x) {
        int L = spell[data[i] * SPW + TSTEPS];
        slen[i] = L;
        atomicAdd(&hist[L], 1);
    }
    __syncthreads();
    if (tid == 0) {
        int off = 0;
        for (int L = TSTEPS; L >= 0; --L) { start[L] = off; off += hist[L]; }
        for (int i = 0; i < BATCH; ++i) perm[start[slen[i]]++] = i;
    }
}

// ---------------------------------------------------------------------------
// gru_step: one GRU time step, fused input+recurrent GEMMs via bf16 WMMA.
// Block = 256 threads (8 waves). Block tile: 32 batch rows x 128 hidden cols.
// Wave tile: 32x16 output (2 M-subtiles sharing B fragments).
// Accumulators zero-init (folds to WMMA inline SRC2=0); biases applied in the
// epilogue. K loop kept as a real loop (unroll 1) with disjoint ping/pong
// fragment register sets: next chunk's loads issue while current chunk's 6
// WMMAs execute, giving a one-iteration load->use distance without copies.
// ---------------------------------------------------------------------------
__global__ __launch_bounds__(256) void s2v_gru_step(
    const int* __restrict__ data, const int* __restrict__ spell,
    const __bf16* __restrict__ Whh,   // [1536,512] bf16 row-major
    const __bf16* __restrict__ Wih,   // [1536,64]
    const __bf16* __restrict__ embb,  // [128,64]
    const float* __restrict__ b_ih, const float* __restrict__ b_hh,
    const float* __restrict__ h_in, const __bf16* __restrict__ hb_in,
    float* __restrict__ h_out, __bf16* __restrict__ hb_out,
    int t)
{
    __shared__ __bf16 sh[MTILE * LDSH_STRIDE]; // h tile, bf16
    __shared__ __bf16 sx[MTILE * LDSX_STRIDE]; // x_t tile (embedded chars)
    __shared__ int slen[MTILE];

    const int tid  = threadIdx.x;
    const int b0   = blockIdx.x * MTILE;                // batch tile base
    const int wave = tid >> 5;
    const int lane = tid & 31;
    const int j0   = blockIdx.y * 128 + wave * 16;      // hidden-col tile base

    // ---- stage h tile: 32 rows x 512 bf16 = 2048 x 16B chunks
    for (int c = tid; c < MTILE * 64; c += 256) {
        const int r = c >> 6, s = c & 63;
        uint4 val = *(const uint4*)(hb_in + (size_t)(b0 + r) * HDIM + s * 8);
        *(uint4*)(&sh[r * LDSH_STRIDE + s * 8]) = val;
    }
    // ---- stage chars/lengths + embedded rows
    if (tid < MTILE) {
        const int w = data[b0 + tid];
        slen[tid] = spell[w * SPW + TSTEPS];
        const int ch = spell[w * SPW + t];
        const uint4* er = (const uint4*)(embb + ch * EDIM);
        uint4* dr = (uint4*)(&sx[tid * LDSX_STRIDE]);
#pragma unroll
        for (int k = 0; k < 8; ++k) dr[k] = er[k];
    }
    __syncthreads();

    const int hi = lane >> 4;   // lane group (0: lanes 0-15, 1: lanes 16-31)
    const int mr = lane & 15;   // M within subtile for A / N for B,C,D

    // zero accumulators: first WMMA of each chain folds to inline SRC2 = 0
    v8f accR0 = {}, accZ0 = {}, accXN0 = {}, accHN0 = {};
    v8f accR1 = {}, accZ1 = {}, accXN1 = {}, accHN1 = {};

    // W_hh rows for this lane's output column (r / z / n gate blocks)
    const __bf16* wr = Whh + (size_t)(j0 + mr) * HDIM;
    const __bf16* wz = Whh + (size_t)(HDIM + j0 + mr) * HDIM;
    const __bf16* wn = Whh + (size_t)(2 * HDIM + j0 + mr) * HDIM;

    // fragment loaders (A from LDS tile, B from weight rows in L2)
    auto loadA = [&](const __bf16* lds, int ldsStride, int kc,
                     Frag& a0, Frag& a1) {
        const int ka = kc + hi * 8;
        a0.q[0] = *(const uint4*)(lds + mr * ldsStride + ka);
        a0.q[1] = *(const uint4*)(lds + mr * ldsStride + ka + 16);
        a1.q[0] = *(const uint4*)(lds + (mr + 16) * ldsStride + ka);
        a1.q[1] = *(const uint4*)(lds + (mr + 16) * ldsStride + ka + 16);
    };
    auto loadB = [&](const __bf16* r0, const __bf16* r1, const __bf16* r2,
                     int kc, Frag& f0, Frag& f1, Frag& f2) {
        const int kb = kc + hi * 8;
        f0.q[0] = *(const uint4*)(r0 + kb);
        f0.q[1] = *(const uint4*)(r0 + kb + 16);
        f1.q[0] = *(const uint4*)(r1 + kb);
        f1.q[1] = *(const uint4*)(r1 + kb + 16);
        f2.q[0] = *(const uint4*)(r2 + kb);
        f2.q[1] = *(const uint4*)(r2 + kb + 16);
    };

    // ping/pong fragment register sets
    Frag a0A, a1A, frA, fzA, fnA;
    Frag a0B, a1B, frB, fzB, fnB;

    // ---- recurrent contraction: K = 512 over h_{t-1}, 2-stage pipeline
    loadA(sh, LDSH_STRIDE, 0, a0A, a1A);
    loadB(wr, wz, wn, 0, frA, fzA, fnA);
#pragma unroll 1
    for (int kc = 0; kc < HDIM; kc += 64) {
        // prefetch odd chunk into B set, compute even chunk from A set
        loadA(sh, LDSH_STRIDE, kc + 32, a0B, a1B);
        loadB(wr, wz, wn, kc + 32, frB, fzB, fnB);
        accR0  = WMMA_BF16(a0A, frA, accR0);
        accR1  = WMMA_BF16(a1A, frA, accR1);
        accZ0  = WMMA_BF16(a0A, fzA, accZ0);
        accZ1  = WMMA_BF16(a1A, fzA, accZ1);
        accHN0 = WMMA_BF16(a0A, fnA, accHN0);
        accHN1 = WMMA_BF16(a1A, fnA, accHN1);
        // prefetch next even chunk into A set, compute odd chunk from B set
        if (kc + 64 < HDIM) {
            loadA(sh, LDSH_STRIDE, kc + 64, a0A, a1A);
            loadB(wr, wz, wn, kc + 64, frA, fzA, fnA);
        }
        accR0  = WMMA_BF16(a0B, frB, accR0);
        accR1  = WMMA_BF16(a1B, frB, accR1);
        accZ0  = WMMA_BF16(a0B, fzB, accZ0);
        accZ1  = WMMA_BF16(a1B, fzB, accZ1);
        accHN0 = WMMA_BF16(a0B, fnB, accHN0);
        accHN1 = WMMA_BF16(a1B, fnB, accHN1);
    }

    // ---- input contraction: K = 64 over x_t (goes to r, z, and xn)
    const __bf16* vr = Wih + (size_t)(j0 + mr) * EDIM;
    const __bf16* vz = Wih + (size_t)(HDIM + j0 + mr) * EDIM;
    const __bf16* vn = Wih + (size_t)(2 * HDIM + j0 + mr) * EDIM;

    loadA(sx, LDSX_STRIDE, 0, a0A, a1A);
    loadB(vr, vz, vn, 0, frA, fzA, fnA);
    loadA(sx, LDSX_STRIDE, 32, a0B, a1B);
    loadB(vr, vz, vn, 32, frB, fzB, fnB);

    // issue epilogue operand loads early so they overlap the final WMMAs
    const float bR  = b_ih[j0 + mr]            + b_hh[j0 + mr];
    const float bZ  = b_ih[HDIM + j0 + mr]     + b_hh[HDIM + j0 + mr];
    const float bXN = b_ih[2 * HDIM + j0 + mr];
    const float bHN = b_hh[2 * HDIM + j0 + mr];
    float hold0[8], hold1[8];
#pragma unroll
    for (int v = 0; v < 8; ++v) {
        const int m = v + hi * 8;
        hold0[v] = h_in[(size_t)(b0 + m) * HDIM + j0 + mr];
        hold1[v] = h_in[(size_t)(b0 + 16 + m) * HDIM + j0 + mr];
    }

    accR0  = WMMA_BF16(a0A, frA, accR0);
    accR1  = WMMA_BF16(a1A, frA, accR1);
    accZ0  = WMMA_BF16(a0A, fzA, accZ0);
    accZ1  = WMMA_BF16(a1A, fzA, accZ1);
    accXN0 = WMMA_BF16(a0A, fnA, accXN0);
    accXN1 = WMMA_BF16(a1A, fnA, accXN1);
    accR0  = WMMA_BF16(a0B, frB, accR0);
    accR1  = WMMA_BF16(a1B, frB, accR1);
    accZ0  = WMMA_BF16(a0B, fzB, accZ0);
    accZ1  = WMMA_BF16(a1B, fzB, accZ1);
    accXN0 = WMMA_BF16(a0B, fnB, accXN0);
    accXN1 = WMMA_BF16(a1B, fnB, accXN1);

    // ---- GRU epilogue (biases folded in here).
    // C/D layout: vgpr v -> M = v + 8*hi, lane&15 -> N.
#pragma unroll
    for (int v = 0; v < 8; ++v) {
        const int m = v + hi * 8;
        {   // row group 0 (rows b0 .. b0+15)
            const size_t o = (size_t)(b0 + m) * HDIM + j0 + mr;
            const float r = 1.0f / (1.0f + __expf(-(accR0[v] + bR)));
            const float z = 1.0f / (1.0f + __expf(-(accZ0[v] + bZ)));
            const float n = tanhf(accXN0[v] + bXN + r * (accHN0[v] + bHN));
            const float hnew = (1.0f - z) * n + z * hold0[v];
            const float hw = (t < slen[m]) ? hnew : hold0[v];
            h_out[o] = hw;
            hb_out[o] = (__bf16)hw;
        }
        {   // row group 1 (rows b0+16 .. b0+31)
            const size_t o = (size_t)(b0 + 16 + m) * HDIM + j0 + mr;
            const float r = 1.0f / (1.0f + __expf(-(accR1[v] + bR)));
            const float z = 1.0f / (1.0f + __expf(-(accZ1[v] + bZ)));
            const float n = tanhf(accXN1[v] + bXN + r * (accHN1[v] + bHN));
            const float hnew = (1.0f - z) * n + z * hold1[v];
            const float hw = (t < slen[16 + m]) ? hnew : hold1[v];
            h_out[o] = hw;
            hb_out[o] = (__bf16)hw;
        }
    }
}

// ---------------------------------------------------------------------------
// finalize: out[i] = h[perm[perm[i]]]  (reference applies perm twice)
// ---------------------------------------------------------------------------
__global__ __launch_bounds__(256) void s2v_finalize(
    const float* __restrict__ hfin, const int* __restrict__ perm,
    float* __restrict__ out)
{
    const int idx = blockIdx.x * blockDim.x + threadIdx.x;
    if (idx >= BATCH * HDIM) return;
    const int i = idx >> 9;
    const int j = idx & 511;
    const int p = perm[perm[i]];
    out[idx] = hfin[(size_t)p * HDIM + j];
}

// ---------------------------------------------------------------------------
extern "C" void kernel_launch(void* const* d_in, const int* in_sizes, int n_in,
                              void* d_out, int out_size, void* d_ws, size_t ws_size,
                              hipStream_t stream) {
    const int*   data  = (const int*)d_in[0];
    const int*   spell = (const int*)d_in[1];   // V x 21 int32
    const float* emb_f = (const float*)d_in[2]; // 128 x 64
    const float* Wih_f = (const float*)d_in[3]; // 1536 x 64
    const float* Whh_f = (const float*)d_in[4]; // 1536 x 512
    const float* b_ih  = (const float*)d_in[5];
    const float* b_hh  = (const float*)d_in[6];
    float* out = (float*)d_out;

    char* base = (char*)d_ws;
    size_t off = 0;
    auto take = [&](size_t bytes) -> char* {
        char* p = base + off;
        off = (off + bytes + 255) & ~(size_t)255;
        return p;
    };
    __bf16* Whh  = (__bf16*)take((size_t)G3H * HDIM * 2);
    __bf16* Wih  = (__bf16*)take((size_t)G3H * EDIM * 2);
    __bf16* embb = (__bf16*)take((size_t)128 * EDIM * 2);
    float*  hf[2]; __bf16* hb[2];
    hf[0] = (float*) take((size_t)BATCH * HDIM * 4);
    hf[1] = (float*) take((size_t)BATCH * HDIM * 4);
    hb[0] = (__bf16*)take((size_t)BATCH * HDIM * 2);
    hb[1] = (__bf16*)take((size_t)BATCH * HDIM * 2);
    int* perm = (int*)take((size_t)BATCH * 4);

    s2v_prep<<<512, 256, 0, stream>>>(Whh_f, Wih_f, emb_f, Whh, Wih, embb,
                                      hf[0], hb[0], hf[1], hb[1]);
    s2v_perm<<<1, 256, 0, stream>>>(data, spell, perm);

    dim3 grid(BATCH / MTILE, HDIM / 128);
    for (int t = 0; t < TSTEPS; ++t) {
        const int in  = t & 1;
        const int outp = 1 - in;
        s2v_gru_step<<<grid, 256, 0, stream>>>(
            data, spell, Whh, Wih, embb, b_ih, b_hh,
            hf[in], hb[in], hf[outp], hb[outp], t);
    }
    // after t = 19 (odd), final h is in buffer 0
    s2v_finalize<<<(BATCH * HDIM + 255) / 256, 256, 0, stream>>>(hf[0], perm, out);
}
